// GATODE_53412213293716
// MI455X (gfx1250) — matrix operand order repaired and checked by reading, not compiled, not run on hardware
//
#include <hip/hip_runtime.h>
#include <stdint.h>

// ---------------------------------------------------------------------------
// GAT (3 layers) + RK4 ODE block for MI455X (gfx1250, wave32, WMMA).
// Dense GEMMs: v_wmma_f32_16x16x32_bf16 (bf16 in, fp32 accumulate), with the
// weight strip staged into LDS via global_load_async_to_lds_b128 (ASYNCcnt).
// ---------------------------------------------------------------------------

typedef __attribute__((ext_vector_type(16))) __bf16 v16bf;
typedef __attribute__((ext_vector_type(8)))  float  v8f;

static constexpr int   kN         = 50000;
static constexpr int   kE         = 850000;
static constexpr int   kH         = 4;
static constexpr int   kK         = 256;     // inner dim of every GEMM
static constexpr float kSlopeAttn = 0.2f;
static constexpr float kSlopeAct  = 0.01f;
static constexpr int   kOdeSteps  = 10;

// ----------------------------- helpers -------------------------------------

__device__ __forceinline__ __bf16 f2bf(float x) {
  unsigned u = __float_as_uint(x);
  unsigned r = u + 0x7FFFu + ((u >> 16) & 1u);   // round-to-nearest-even
  unsigned short h = (unsigned short)(r >> 16);
  __bf16 out;
  __builtin_memcpy(&out, &h, 2);
  return out;
}

// monotonic float<->uint map so unsigned atomicMax == float max
__device__ __forceinline__ unsigned fenc(float x) {
  unsigned u = __float_as_uint(x);
  return (u & 0x80000000u) ? ~u : (u | 0x80000000u);
}
__device__ __forceinline__ float fdec(unsigned e) {
  unsigned u = (e & 0x80000000u) ? (e & 0x7FFFFFFFu) : ~e;
  return __uint_as_float(u);
}
#define ENC_NEG_INF 0x007FFFFFu   // fenc(-inf)

// ----------------------------- small kernels --------------------------------

__global__ void k_f32_to_bf16(const float* __restrict__ in, __bf16* __restrict__ out, int n) {
  int i = blockIdx.x * blockDim.x + threadIdx.x;
  if (i < n) out[i] = f2bf(in[i]);
}

__global__ void k_lrelu_to_bf16(const float* __restrict__ in, __bf16* __restrict__ out, int n) {
  int i = blockIdx.x * blockDim.x + threadIdx.x;
  if (i < n) {
    float v = in[i];
    v = v > 0.f ? v : kSlopeAct * v;
    out[i] = f2bf(v);
  }
}

// W[K][C] (f32, row-major) -> WT[C][K] (bf16) so B fragments are contiguous
__global__ void k_wt_bf16(const float* __restrict__ W, __bf16* __restrict__ WT, int K, int C) {
  int i = blockIdx.x * blockDim.x + threadIdx.x;
  if (i >= K * C) return;
  int k = i / C, c = i - k * C;
  WT[(size_t)c * K + k] = f2bf(W[i]);
}

__global__ void k_init_attn(unsigned* __restrict__ emax, float* __restrict__ denom, int n) {
  int i = blockIdx.x * blockDim.x + threadIdx.x;
  if (i < n) { emax[i] = ENC_NEG_INF; denom[i] = 0.f; }
}

__global__ void k_init_out_bias(float* __restrict__ outb, const float* __restrict__ b, int total, int C) {
  int i = blockIdx.x * blockDim.x + threadIdx.x;
  if (i < total) outb[i] = b[i % C];
}

// el[n,h] = <feat[n,h,:], al[h,:]> ; er likewise
__global__ void k_attn_scores(const float* __restrict__ feat, const float* __restrict__ al,
                              const float* __restrict__ ar, float* __restrict__ el,
                              float* __restrict__ er, int N_, int Hh, int Dd) {
  int i = blockIdx.x * blockDim.x + threadIdx.x;
  if (i >= N_ * Hh) return;
  int n = i / Hh, h = i - n * Hh;
  const float4* fp  = (const float4*)(feat + (size_t)n * (Hh * Dd) + h * Dd);
  const float4* alp = (const float4*)(al + h * Dd);
  const float4* arp = (const float4*)(ar + h * Dd);
  float sl = 0.f, sr = 0.f;
  for (int q = 0; q < Dd / 4; ++q) {
    float4 f = fp[q], A = alp[q], R = arp[q];
    sl += f.x * A.x + f.y * A.y + f.z * A.z + f.w * A.w;
    sr += f.x * R.x + f.y * R.y + f.z * R.z + f.w * R.w;
  }
  el[i] = sl;
  er[i] = sr;
}

__device__ __forceinline__ float edge_e(const int* src, const int* dst,
                                        const float* el, const float* er,
                                        int e, int h, int Hh, int& d_out_) {
  int s = src[e], d = dst[e];
  d_out_ = d;
  float x = el[s * Hh + h] + er[d * Hh + h];
  return x > 0.f ? x : kSlopeAttn * x;
}

__global__ void k_edge_max(const int* __restrict__ src, const int* __restrict__ dst,
                           const float* __restrict__ el, const float* __restrict__ er,
                           unsigned* __restrict__ emax, int E_, int Hh) {
  int i = blockIdx.x * blockDim.x + threadIdx.x;
  if (i >= E_ * Hh) return;
  int e = i / Hh, h = i - e * Hh, d;
  float x = edge_e(src, dst, el, er, e, h, Hh, d);
  atomicMax(&emax[d * Hh + h], fenc(x));
}

__global__ void k_edge_sum(const int* __restrict__ src, const int* __restrict__ dst,
                           const float* __restrict__ el, const float* __restrict__ er,
                           const unsigned* __restrict__ emax, float* __restrict__ denom,
                           int E_, int Hh) {
  int i = blockIdx.x * blockDim.x + threadIdx.x;
  if (i >= E_ * Hh) return;
  int e = i / Hh, h = i - e * Hh, d;
  float x = edge_e(src, dst, el, er, e, h, Hh, d);
  atomicAdd(&denom[d * Hh + h], __expf(x - fdec(emax[d * Hh + h])));
}

// 16-way split per (edge, head): one float4 + 4 atomics per thread (D = 64).
__global__ void k_edge_aggregate(const int* __restrict__ src, const int* __restrict__ dst,
                                 const float* __restrict__ el, const float* __restrict__ er,
                                 const unsigned* __restrict__ emax, const float* __restrict__ denom,
                                 const float* __restrict__ feat, float* __restrict__ outb,
                                 int E_, int Hh) {
  int i = blockIdx.x * blockDim.x + threadIdx.x;
  if (i >= E_ * Hh * 16) return;
  int q  = i & 15;
  int eh = i >> 4;
  int e = eh / Hh, h = eh - e * Hh;
  int s = src[e], d = dst[e];
  int idx = d * Hh + h;
  float x = el[s * Hh + h] + er[idx];
  x = x > 0.f ? x : kSlopeAttn * x;
  float alpha = __expf(x - fdec(emax[idx])) / denom[idx];
  const float4 m = *(const float4*)(feat + (size_t)s * (Hh * 64) + h * 64 + q * 4);
  float* op = outb + (size_t)d * (Hh * 64) + h * 64 + q * 4;
  atomicAdd(op + 0, m.x * alpha);
  atomicAdd(op + 1, m.y * alpha);
  atomicAdd(op + 2, m.z * alpha);
  atomicAdd(op + 3, m.w * alpha);
}

// ----------------------------- WMMA GEMM ------------------------------------
// C[M,Ncols] = A[M,256](bf16) * WT[Ncols,256](bf16)^T, fp32 out, no bias.
// Block = 8 waves, one 64-column strip of WT async-staged into LDS and shared
// by all waves; each wave computes a 16x64 strip (A fragment reused 4x).
// LDS row padded to 264 bf16 (132 dwords, = 4 mod 64 banks) so B-fragment
// ds_load_b128 across lanes 0..15 is bank-conflict free.
#define BS_PAD 264

__global__ void __launch_bounds__(256)
k_gemm_bf16_wmma(const __bf16* __restrict__ A,
                 const __bf16* __restrict__ WT,
                 float* __restrict__ C,
                 int Mtiles, int Ncols) {
  __shared__ __bf16 Bs[64][BS_PAD];
  const int strip = blockIdx.y;
  const int tid = threadIdx.x;

  // ---- async-stage this strip's 64 WT rows (64 x 256 bf16 = 32KB) ----------
  {
    const char* gbase = (const char*)(WT + (size_t)strip * 64 * kK);
#pragma unroll
    for (int i = 0; i < 8; ++i) {
      int chunk = tid + i * 256;                   // 2048 x 16B chunks
      int row = chunk >> 5;
      int off = (chunk & 31) * 16;                 // byte offset within row
      unsigned lds = (unsigned)(uintptr_t)(&Bs[0][0]) + (unsigned)(row * (BS_PAD * 2) + off);
      unsigned long long gaddr =
          (unsigned long long)(uintptr_t)(gbase + (size_t)row * (kK * 2) + off);
      asm volatile("global_load_async_to_lds_b128 %0, %1, off"
                   :: "v"(lds), "v"(gaddr) : "memory");
    }
    asm volatile("s_wait_asynccnt 0" ::: "memory");
  }
  __syncthreads();

  const int wv = tid >> 5;
  const int mTile = blockIdx.x * 8 + wv;
  const int lane = tid & 31;
  const int half = lane >> 4;                      // 0: lanes 0-15, 1: lanes 16-31
  const int r = lane & 15;
  if (mTile >= Mtiles) return;                     // wave-granular, after barrier

  const __bf16* ap = A + (size_t)(mTile * 16 + r) * kK + half * 8;
  v8f acc[4] = {};
  for (int kk = 0; kk < kK; kk += 32) {
    union { v16bf v; uint4 q[2]; } a;
    a.q[0] = *(const uint4*)(ap + kk);             // K kk+half*8    .. +7
    a.q[1] = *(const uint4*)(ap + kk + 16);        // K kk+half*8+16 .. +7
#pragma unroll
    for (int nt = 0; nt < 4; ++nt) {
      union { v16bf v; uint4 q[2]; } b;
      const __bf16* bp = &Bs[nt * 16 + r][kk + half * 8];
      b.q[0] = *(const uint4*)(bp);
      b.q[1] = *(const uint4*)(bp + 16);
      acc[nt] = __builtin_amdgcn_wmma_f32_16x16x32_bf16(false, a.v, false, b.v,
                                                        (short)0, acc[nt], false, false);
    }
  }
  float* cp0 = C + (size_t)(mTile * 16 + half * 8) * Ncols + strip * 64 + r;
#pragma unroll
  for (int nt = 0; nt < 4; ++nt)
#pragma unroll
    for (int j = 0; j < 8; ++j)
      cp0[(size_t)j * Ncols + nt * 16] = acc[nt][j];
}

// ----------------------------- fused RK4 ODE --------------------------------
// y_{t+1} via classic RK4 on f(y) = y @ Wf^T + bf, 10 steps, fully fused.
// One wave per 16-row tile; Wf B-fragments live in registers for all 320 WMMAs;
// y / k / ksum are register tiles in C-layout; only the u -> A-fragment
// transpose round-trips through LDS (guarded by s_wait_dscnt).
__global__ void __launch_bounds__(32)
k_ode_rk4_wmma(const float* __restrict__ h0,      // [M,64] fp32
               const __bf16* __restrict__ Wfb,    // [64,64] bf16, row-major (row = out col)
               const float* __restrict__ bvec,    // [64]
               float* __restrict__ out, int M) {
  __shared__ float u[16][72];                     // padded to dodge bank conflicts
  const int lane = threadIdx.x & 31;
  const int half = lane >> 4;
  const int r = lane & 15;
  const int rowBase = blockIdx.x * 16;
  if (rowBase >= M) return;

  union BF { v16bf v; uint4 q[2]; };
  BF Bf[4][2];
#pragma unroll
  for (int nt = 0; nt < 4; ++nt) {
    const __bf16* bp = Wfb + (size_t)(nt * 16 + r) * 64 + half * 8;
#pragma unroll
    for (int ks = 0; ks < 2; ++ks) {
      Bf[nt][ks].q[0] = *(const uint4*)(bp + ks * 32);
      Bf[nt][ks].q[1] = *(const uint4*)(bp + ks * 32 + 16);
    }
  }
  float bias[4];
#pragma unroll
  for (int nt = 0; nt < 4; ++nt) bias[nt] = bvec[nt * 16 + r];

  v8f y[4];
#pragma unroll
  for (int nt = 0; nt < 4; ++nt)
#pragma unroll
    for (int j = 0; j < 8; ++j)
      y[nt][j] = h0[(size_t)(rowBase + j + half * 8) * 64 + nt * 16 + r];

  const float hstep = 1.0f / kOdeSteps;
  const float cin[3] = {0.5f * hstep, 0.5f * hstep, hstep};
  const float wgt[4] = {1.f, 2.f, 2.f, 1.f};

  for (int step = 0; step < kOdeSteps; ++step) {
    v8f ksum[4] = {};
    v8f uu[4];
#pragma unroll
    for (int nt = 0; nt < 4; ++nt) uu[nt] = y[nt];

#pragma unroll
    for (int s = 0; s < 4; ++s) {
      // u tile -> LDS (C-layout scatter)
#pragma unroll
      for (int nt = 0; nt < 4; ++nt)
#pragma unroll
        for (int j = 0; j < 8; ++j)
          u[j + half * 8][nt * 16 + r] = uu[nt][j];
      __asm__ volatile("s_wait_dscnt 0" ::: "memory");   // CDNA5 split DS counter

      // LDS -> A fragments (row-per-lane, on-the-fly bf16 convert)
      union AF { v16bf v; __bf16 e[16]; } af[2];
#pragma unroll
      for (int ks = 0; ks < 2; ++ks) {
        const int base = ks * 32 + half * 8;
#pragma unroll
        for (int t = 0; t < 8; ++t) {
          af[ks].e[t]     = f2bf(u[r][base + t]);
          af[ks].e[8 + t] = f2bf(u[r][base + 16 + t]);
        }
      }

      v8f kt[4];
#pragma unroll
      for (int nt = 0; nt < 4; ++nt) {
        v8f acc;
#pragma unroll
        for (int j = 0; j < 8; ++j) acc[j] = bias[nt];
        acc = __builtin_amdgcn_wmma_f32_16x16x32_bf16(false, af[0].v, false, Bf[nt][0].v,
                                                      (short)0, acc, false, false);
        acc = __builtin_amdgcn_wmma_f32_16x16x32_bf16(false, af[1].v, false, Bf[nt][1].v,
                                                      (short)0, acc, false, false);
        kt[nt] = acc;
      }
#pragma unroll
      for (int nt = 0; nt < 4; ++nt) {
        ksum[nt] += wgt[s] * kt[nt];
        if (s < 3) uu[nt] = y[nt] + cin[s] * kt[nt];
      }
    }
#pragma unroll
    for (int nt = 0; nt < 4; ++nt) y[nt] += (hstep / 6.0f) * ksum[nt];
  }

#pragma unroll
  for (int nt = 0; nt < 4; ++nt)
#pragma unroll
    for (int j = 0; j < 8; ++j)
      out[(size_t)(rowBase + j + half * 8) * 64 + nt * 16 + r] = y[nt][j];
}

// ----------------------------- orchestration --------------------------------

extern "C" void kernel_launch(void* const* d_in, const int* in_sizes, int n_in,
                              void* d_out, int out_size, void* d_ws, size_t ws_size,
                              hipStream_t stream) {
  (void)in_sizes; (void)out_size; (void)ws_size;
  if (n_in < 17) return;

  const float* x   = (const float*)d_in[0];
  const int*   src = (const int*)d_in[1];
  const int*   dst = (const int*)d_in[2];
  const float* W0  = (const float*)d_in[3];
  const float* al0 = (const float*)d_in[4];
  const float* ar0 = (const float*)d_in[5];
  const float* b0  = (const float*)d_in[6];
  const float* W1  = (const float*)d_in[7];
  const float* al1 = (const float*)d_in[8];
  const float* ar1 = (const float*)d_in[9];
  const float* b1  = (const float*)d_in[10];
  const float* W2  = (const float*)d_in[11];
  const float* al2 = (const float*)d_in[12];
  const float* ar2 = (const float*)d_in[13];
  const float* b2  = (const float*)d_in[14];
  const float* Wf  = (const float*)d_in[15];
  const float* bfv = (const float*)d_in[16];

  char* ws = (char*)d_ws;
  size_t off = 0;
  auto carve = [&](size_t bytes) -> void* {
    void* p = ws + off;
    off = (off + bytes + 255) & ~(size_t)255;
    return p;
  };
  __bf16*   Xbf   = (__bf16*)carve((size_t)kN * 256 * 2);   // bf16 layer input
  float*    feat  = (float*)carve((size_t)kN * 256 * 4);    // x @ W
  float*    outb  = (float*)carve((size_t)kN * 256 * 4);    // aggregated output
  float*    el    = (float*)carve((size_t)kN * kH * 4);
  float*    er    = (float*)carve((size_t)kN * kH * 4);
  unsigned* emax  = (unsigned*)carve((size_t)kN * kH * 4);
  float*    denom = (float*)carve((size_t)kN * kH * 4);
  __bf16*   WT    = (__bf16*)carve((size_t)256 * 256 * 2);
  __bf16*   Wfb   = (__bf16*)carve((size_t)64 * 64 * 2);

  auto cdiv = [](int a, int b) { return (a + b - 1) / b; };
  const int mTiles = kN / 16;                 // 3125

  // input -> bf16 once
  k_f32_to_bf16<<<cdiv(kN * 256, 256), 256, 0, stream>>>(x, Xbf, kN * 256);

  auto run_layer = [&](const float* W, const float* al, const float* ar, const float* bb,
                       int Hh, int Dd, bool act) {
    const int C = Hh * Dd;
    k_wt_bf16<<<cdiv(kK * C, 256), 256, 0, stream>>>(W, WT, kK, C);
    k_gemm_bf16_wmma<<<dim3(cdiv(mTiles, 8), C / 64), 256, 0, stream>>>(Xbf, WT, feat,
                                                                        mTiles, C);
    k_attn_scores<<<cdiv(kN * Hh, 256), 256, 0, stream>>>(feat, al, ar, el, er, kN, Hh, Dd);
    k_init_attn<<<cdiv(kN * Hh, 256), 256, 0, stream>>>(emax, denom, kN * Hh);
    k_init_out_bias<<<cdiv(kN * C, 256), 256, 0, stream>>>(outb, bb, kN * C, C);
    k_edge_max<<<cdiv(kE * Hh, 256), 256, 0, stream>>>(src, dst, el, er, emax, kE, Hh);
    k_edge_sum<<<cdiv(kE * Hh, 256), 256, 0, stream>>>(src, dst, el, er, emax, denom, kE, Hh);
    k_edge_aggregate<<<cdiv(kE * Hh * 16, 256), 256, 0, stream>>>(src, dst, el, er, emax,
                                                                  denom, feat, outb, kE, Hh);
    if (act)
      k_lrelu_to_bf16<<<cdiv(kN * C, 256), 256, 0, stream>>>(outb, Xbf, kN * C);
  };

  run_layer(W0, al0, ar0, b0, 4, 64, true);
  run_layer(W1, al1, ar1, b1, 4, 64, true);
  run_layer(W2, al2, ar2, b2, 1, 64, false);   // outb now holds [N,64] fp32

  k_f32_to_bf16<<<cdiv(64 * 64, 256), 256, 0, stream>>>(Wf, Wfb, 64 * 64);
  k_ode_rk4_wmma<<<mTiles, 32, 0, stream>>>(outb, Wfb, bfv, (float*)d_out, kN);
}
